// MultiHeadSelfAttention_2302102470958
// MI455X (gfx1250) — matrix operand order repaired and checked by reading, not compiled, not run on hardware
//
#include <hip/hip_runtime.h>
#include <hip/hip_bf16.h>

#define DEVINL __device__ __forceinline__

typedef __bf16 v16bf __attribute__((ext_vector_type(16)));
typedef float  v8f   __attribute__((ext_vector_type(8)));

constexpr int BB = 2;
constexpr int TT = 2048;
constexpr int DD = 1024;
constexpr int HH = 16;
constexpr int HD = 64;
constexpr int NN = BB * TT;   // 4096 tokens
constexpr int D3 = 3 * DD;    // 3072

union Frag {
  v16bf v;
  uint4 q[2];
  unsigned short h[16];
};

DEVINL unsigned short f2bf(float f) {
  unsigned int u = __float_as_uint(f);
  unsigned int r = (u + 0x7FFFu + ((u >> 16) & 1u)) >> 16;  // RNE
  return (unsigned short)r;
}

DEVINL v8f vzero() {
  v8f z;
#pragma unroll
  for (int i = 0; i < 8; ++i) z[i] = 0.0f;
  return z;
}

DEVINL v8f wmma_bf16(const Frag& a, const Frag& b, v8f c) {
  return __builtin_amdgcn_wmma_f32_16x16x32_bf16(
      false, a.v, false, b.v, (short)0, c, false, false);
}

// ---------------- convert / transpose kernels ----------------

__global__ void k_f2bf(const float* __restrict__ in,
                       unsigned short* __restrict__ out, int n) {
  int i = blockIdx.x * blockDim.x + threadIdx.x;
  if (i < n) out[i] = f2bf(in[i]);
}

// in: [rows, cols] f32   ->   out: [cols, rows] bf16 (transposed)
__global__ void k_transpose_f2bf(const float* __restrict__ in,
                                 unsigned short* __restrict__ out,
                                 int rows, int cols) {
  int i = blockIdx.x * blockDim.x + threadIdx.x;
  if (i < rows * cols) {
    int o = i / rows;   // out-row == column of in
    int r = i % rows;
    out[i] = f2bf(in[r * cols + o]);
  }
}

// ---------------- GEMM1: qkv = x @ Wqkv + b, scattered to Q/K/Vt ----------------
// one wave computes a 16(M) x 64(N) tile; A fragment reused across 4 B tiles

__global__ __launch_bounds__(32) void k_gemm_qkv(
    const unsigned short* __restrict__ xb,   // [NN, DD] bf16
    const unsigned short* __restrict__ wt,   // [D3, DD] bf16 (W^T)
    const float* __restrict__ bias,          // [D3]
    unsigned short* __restrict__ Qb,         // [B*H, T, HD]
    unsigned short* __restrict__ Kb,         // [B*H, T, HD]
    unsigned short* __restrict__ Vt)         // [B*H, HD, T]
{
  const int lane = threadIdx.x;
  const int col0 = blockIdx.x * 64;
  const int row0 = blockIdx.y * 16;
  const int l15 = lane & 15;
  const int hi  = (lane & 16) ? 1 : 0;

  v8f acc[4];
#pragma unroll
  for (int j = 0; j < 4; ++j) acc[j] = vzero();

  const unsigned short* arow = xb + (size_t)(row0 + l15) * DD + (hi ? 8 : 0);
  for (int kk = 0; kk < DD / 32; ++kk) {
    Frag a;
    const unsigned short* ap = arow + kk * 32;
    a.q[0] = *(const uint4*)ap;
    a.q[1] = *(const uint4*)(ap + 16);
#pragma unroll
    for (int j = 0; j < 4; ++j) {
      Frag b;
      const unsigned short* bp =
          wt + (size_t)(col0 + j * 16 + l15) * DD + kk * 32 + (hi ? 8 : 0);
      b.q[0] = *(const uint4*)bp;
      b.q[1] = *(const uint4*)(bp + 16);
      acc[j] = wmma_bf16(a, b, acc[j]);
    }
  }

#pragma unroll
  for (int j = 0; j < 4; ++j) {
    int col = col0 + j * 16 + l15;
    float bv = bias[col];
    int sec  = col >> 10;        // 0:q 1:k 2:v
    int cc   = col & 1023;
    int head = cc >> 6;
    int dd   = cc & 63;
#pragma unroll
    for (int r = 0; r < 8; ++r) {
      int rowi = row0 + r + hi * 8;
      int b_ = rowi >> 11;       // / T
      int t_ = rowi & 2047;
      int bh = b_ * HH + head;
      unsigned short val = f2bf(acc[j][r] + bv);
      if (sec == 0)      Qb[((size_t)bh * TT + t_) * HD + dd] = val;
      else if (sec == 1) Kb[((size_t)bh * TT + t_) * HD + dd] = val;
      else               Vt[((size_t)bh * HD + dd) * TT + t_] = val;
    }
  }
}

// ---------------- flash attention: one wave = 16 q-rows of one (b,h) ----------------

__global__ __launch_bounds__(256) void k_flash(
    const unsigned short* __restrict__ Qb,
    const unsigned short* __restrict__ Kb,
    const unsigned short* __restrict__ Vt,
    unsigned short* __restrict__ Ob)   // [NN, DD] bf16, token-major
{
  __shared__ __align__(16) unsigned short pst[8][16 * 32];
  const int lane = threadIdx.x & 31;
  const int wave = threadIdx.x >> 5;
  const int bh = blockIdx.y;
  const int q0 = (blockIdx.x * 8 + wave) * 16;
  const int l15 = lane & 15;
  const int hi  = (lane & 16) ? 1 : 0;

  // Q A-fragments (d = 0..31, 32..63) held in registers for entire kv loop
  Frag aq[2];
#pragma unroll
  for (int kk = 0; kk < 2; ++kk) {
    const unsigned short* qp =
        Qb + ((size_t)bh * TT + q0 + l15) * HD + kk * 32 + (hi ? 8 : 0);
    aq[kk].q[0] = *(const uint4*)qp;
    aq[kk].q[1] = *(const uint4*)(qp + 16);
  }

  v8f o[4];
#pragma unroll
  for (int dt = 0; dt < 4; ++dt) o[dt] = vzero();
  float m[8], l[8];
#pragma unroll
  for (int r = 0; r < 8; ++r) { m[r] = -1e30f; l[r] = 0.0f; }

  for (int kb = 0; kb < TT / 32; ++kb) {
    const int kbase = kb * 32;
    // S tile: 16 q x 32 k  (two 16x16 C fragments)
    v8f s[2];
    s[0] = vzero(); s[1] = vzero();
#pragma unroll
    for (int nt = 0; nt < 2; ++nt) {
#pragma unroll
      for (int kk = 0; kk < 2; ++kk) {
        Frag bk;  // B[d,k] = K[kcol,d] : contiguous along d per lane
        const unsigned short* kp =
            Kb + ((size_t)bh * TT + kbase + nt * 16 + l15) * HD + kk * 32 + (hi ? 8 : 0);
        bk.q[0] = *(const uint4*)kp;
        bk.q[1] = *(const uint4*)(kp + 16);
        s[nt] = wmma_bf16(aq[kk], bk, s[nt]);
      }
    }

    // online softmax (row = shared across a 16-lane half, per C-VGPR r)
    float p0[8], p1[8];
#pragma unroll
    for (int r = 0; r < 8; ++r) {
      float s0 = s[0][r] * 0.125f;   // 1/sqrt(64)
      float s1 = s[1][r] * 0.125f;
      float t = fmaxf(s0, s1);
      t = fmaxf(t, __shfl_xor(t, 1, 32));
      t = fmaxf(t, __shfl_xor(t, 2, 32));
      t = fmaxf(t, __shfl_xor(t, 4, 32));
      t = fmaxf(t, __shfl_xor(t, 8, 32));
      float mn = fmaxf(m[r], t);
      float alpha = __expf(m[r] - mn);
      p0[r] = __expf(s0 - mn);
      p1[r] = __expf(s1 - mn);
      float rs = p0[r] + p1[r];
      rs += __shfl_xor(rs, 1, 32);
      rs += __shfl_xor(rs, 2, 32);
      rs += __shfl_xor(rs, 4, 32);
      rs += __shfl_xor(rs, 8, 32);
      l[r] = l[r] * alpha + rs;
      m[r] = mn;
#pragma unroll
      for (int dt = 0; dt < 4; ++dt) o[dt][r] *= alpha;
    }

    // reshape P (C layout) -> A layout via wave-private LDS (DS ops in-order per wave)
    unsigned short* ps = pst[wave];
#pragma unroll
    for (int r = 0; r < 8; ++r) {
      int row = r + hi * 8;
      ps[row * 32 + l15]      = f2bf(p0[r]);
      ps[row * 32 + 16 + l15] = f2bf(p1[r]);
    }
    Frag pf;
    const unsigned short* pp = ps + l15 * 32 + (hi ? 8 : 0);
    pf.q[0] = *(const uint4*)pp;
    pf.q[1] = *(const uint4*)(pp + 16);

    // O += P @ V   (Vt: B[k,d] contiguous along k per lane)
#pragma unroll
    for (int dt = 0; dt < 4; ++dt) {
      Frag bv;
      const unsigned short* vp =
          Vt + ((size_t)bh * HD + dt * 16 + l15) * TT + kbase + (hi ? 8 : 0);
      bv.q[0] = *(const uint4*)vp;
      bv.q[1] = *(const uint4*)(vp + 16);
      o[dt] = wmma_bf16(pf, bv, o[dt]);
    }
  }

  // normalize + scatter to token-major [NN, DD]
  const int b_ = bh >> 4;
  const int head = bh & 15;
#pragma unroll
  for (int r = 0; r < 8; ++r) {
    float inv = 1.0f / l[r];
    size_t n = (size_t)b_ * TT + q0 + r + hi * 8;
#pragma unroll
    for (int dt = 0; dt < 4; ++dt) {
      int col = head * HD + dt * 16 + l15;
      Ob[n * DD + col] = f2bf(o[dt][r] * inv);
    }
  }
}

// ---------------- GEMM2: out = attnO @ Wout + b (f32 output) ----------------

__global__ __launch_bounds__(32) void k_gemm_out(
    const unsigned short* __restrict__ ob,   // [NN, DD] bf16
    const unsigned short* __restrict__ wt,   // [DD, DD] bf16 (W^T)
    const float* __restrict__ bias,          // [DD]
    float* __restrict__ out)                 // [NN, DD] f32
{
  const int lane = threadIdx.x;
  const int col0 = blockIdx.x * 64;
  const int row0 = blockIdx.y * 16;
  const int l15 = lane & 15;
  const int hi  = (lane & 16) ? 1 : 0;

  v8f acc[4];
#pragma unroll
  for (int j = 0; j < 4; ++j) acc[j] = vzero();

  const unsigned short* arow = ob + (size_t)(row0 + l15) * DD + (hi ? 8 : 0);
  for (int kk = 0; kk < DD / 32; ++kk) {
    Frag a;
    const unsigned short* ap = arow + kk * 32;
    a.q[0] = *(const uint4*)ap;
    a.q[1] = *(const uint4*)(ap + 16);
#pragma unroll
    for (int j = 0; j < 4; ++j) {
      Frag b;
      const unsigned short* bp =
          wt + (size_t)(col0 + j * 16 + l15) * DD + kk * 32 + (hi ? 8 : 0);
      b.q[0] = *(const uint4*)bp;
      b.q[1] = *(const uint4*)(bp + 16);
      acc[j] = wmma_bf16(a, b, acc[j]);
    }
  }

#pragma unroll
  for (int j = 0; j < 4; ++j) {
    int col = col0 + j * 16 + l15;
    float bv = bias[col];
#pragma unroll
    for (int r = 0; r < 8; ++r) {
      int rowi = row0 + r + hi * 8;
      out[(size_t)rowi * DD + col] = acc[j][r] + bv;
    }
  }
}

// ---------------- host launcher ----------------

extern "C" void kernel_launch(void* const* d_in, const int* in_sizes, int n_in,
                              void* d_out, int out_size, void* d_ws, size_t ws_size,
                              hipStream_t stream) {
  (void)in_sizes; (void)n_in; (void)out_size; (void)ws_size;
  const float* x    = (const float*)d_in[0];
  const float* Wqkv = (const float*)d_in[1];
  const float* bqkv = (const float*)d_in[2];
  const float* Wout = (const float*)d_in[3];
  const float* bout = (const float*)d_in[4];
  float* out = (float*)d_out;

  char* ws = (char*)d_ws;
  size_t off = 0;
  auto carve = [&](size_t bytes) {
    void* p = ws + off;
    off += (bytes + 255) & ~(size_t)255;
    return p;
  };
  unsigned short* xb  = (unsigned short*)carve((size_t)NN * DD * 2);
  unsigned short* wqt = (unsigned short*)carve((size_t)D3 * DD * 2);
  unsigned short* wot = (unsigned short*)carve((size_t)DD * DD * 2);
  unsigned short* Qb  = (unsigned short*)carve((size_t)BB * HH * TT * HD * 2);
  unsigned short* Kb  = (unsigned short*)carve((size_t)BB * HH * TT * HD * 2);
  unsigned short* Vt  = (unsigned short*)carve((size_t)BB * HH * TT * HD * 2);
  unsigned short* Ob  = (unsigned short*)carve((size_t)NN * DD * 2);

  k_f2bf<<<(NN * DD + 255) / 256, 256, 0, stream>>>(x, xb, NN * DD);
  k_transpose_f2bf<<<(DD * D3 + 255) / 256, 256, 0, stream>>>(Wqkv, wqt, DD, D3);
  k_transpose_f2bf<<<(DD * DD + 255) / 256, 256, 0, stream>>>(Wout, wot, DD, DD);

  k_gemm_qkv<<<dim3(D3 / 64, NN / 16), 32, 0, stream>>>(xb, wqt, bqkv, Qb, Kb, Vt);
  k_flash<<<dim3(TT / 16 / 8, BB * HH), 256, 0, stream>>>(Qb, Kb, Vt, Ob);
  k_gemm_out<<<dim3(DD / 64, NN / 16), 32, 0, stream>>>(Ob, wot, bout, out);
}
